// GroupedQueryAttention_438086664772
// MI455X (gfx1250) — matrix operand order repaired and checked by reading, not compiled, not run on hardware
//
#include <hip/hip_runtime.h>
#include <hip/hip_bf16.h>

// ---------------------------------------------------------------------------
// GQA attention block for MI455X (gfx1250, wave32, WMMA 16x16x32 f16)
//   B=2, S=2048, E=2048, H=32, KV=8, D=64, G=4
// ---------------------------------------------------------------------------

typedef __attribute__((ext_vector_type(16))) _Float16 v16h;
typedef __attribute__((ext_vector_type(8)))  _Float16 v8h;
typedef __attribute__((ext_vector_type(8)))  float    v8f;

union AFrag { v16h v; v8h seg[2]; _Float16 h[16]; };

constexpr int Bc  = 2;
constexpr int Sc  = 2048;
constexpr int Ec  = 2048;
constexpr int Hc  = 32;
constexpr int KVc = 8;
constexpr int Dc  = 64;

__device__ __forceinline__ v8f wmma_f16(v16h a, v16h b, v8f c) {
  // D = A(16x32 f16) * B(32x16 f16) + C(16x16 f32)
  return __builtin_amdgcn_wmma_f32_16x16x32_f16(
      /*neg_a=*/false, a, /*neg_b=*/false, b,
      /*c_mod=*/(short)0, c, /*reuse_a=*/false, /*reuse_b=*/false);
}

// ---------------------------------------------------------------------------
// Elementwise f32 -> f16 convert (grid-stride)
// ---------------------------------------------------------------------------
__global__ void cvt_f16_kernel(const float* __restrict__ in,
                               _Float16* __restrict__ out, long n) {
  long i = (long)blockIdx.x * blockDim.x + threadIdx.x;
  long stride = (long)gridDim.x * blockDim.x;
  for (; i < n; i += stride) out[i] = (_Float16)in[i];
}

// ---------------------------------------------------------------------------
// Transpose + convert: out[c][r] (f16, [cols][rows]) = in[r][c] (f32)
// LDS-tiled 32x32, block (32,8). rows, cols multiples of 32.
// ---------------------------------------------------------------------------
__global__ __launch_bounds__(256)
void transpose_cvt_kernel(const float* __restrict__ in,
                          _Float16* __restrict__ out, int rows, int cols) {
  __shared__ _Float16 tile[32][33];
  const int tx = threadIdx.x, ty = threadIdx.y;
  const int c0 = blockIdx.x * 32, r0 = blockIdx.y * 32;
#pragma unroll
  for (int i = 0; i < 32; i += 8)
    tile[ty + i][tx] = (_Float16)in[(size_t)(r0 + ty + i) * cols + c0 + tx];
  __syncthreads();
#pragma unroll
  for (int i = 0; i < 32; i += 8)
    out[(size_t)(c0 + ty + i) * rows + r0 + tx] = tile[tx][ty + i];
}

// ---------------------------------------------------------------------------
// GEMM: C[m][n] = sum_k A[m][k] * Bt[n][k] (+ bias[n])
// A: f16 [M][K] row-major, Bt: f16 [N][K] row-major, C: f32 [M][N].
// Block = 128 threads = 4 waves; wave tile 32(M) x 64(N), K-step 32:
// per K-step a wave does 4 A-frag + 8 B-frag b128 loads for 8 WMMAs
// (B-fragments reused across the two M sub-tiles -> 1.5 loads/WMMA).
// M % 32 == 0, N % 256 == 0, K % 32 == 0.
// ---------------------------------------------------------------------------
__global__ __launch_bounds__(128)
void gemm_f16_wmma_kernel(const _Float16* __restrict__ A,
                          const _Float16* __restrict__ Bt,
                          const float* __restrict__ bias,
                          float* __restrict__ C, int M, int N, int K) {
  const int lane = threadIdx.x & 31;
  const int wave = threadIdx.x >> 5;
  const int hi   = lane >> 4;   // half-wave select
  const int l16  = lane & 15;
  const int m0 = blockIdx.y * 32;
  const int n0 = blockIdx.x * 256 + wave * 64;

  v8f acc[2][4] = {};
  const _Float16* arow0 = A + (size_t)(m0 + l16) * K;
  const _Float16* arow1 = A + (size_t)(m0 + 16 + l16) * K;

  for (int k = 0; k < K; k += 32) {
    // A-frag: lane<16 rows 0-15 K{0..15}; lane>=16 K{16..31} (matches B)
    AFrag a0, a1;
    a0.seg[0] = *(const v8h*)(arow0 + k + 16 * hi);
    a0.seg[1] = *(const v8h*)(arow0 + k + 16 * hi + 8);
    a1.seg[0] = *(const v8h*)(arow1 + k + 16 * hi);
    a1.seg[1] = *(const v8h*)(arow1 + k + 16 * hi + 8);
#pragma unroll
    for (int j = 0; j < 4; ++j) {
      // B-frag: lane = column n, 16 sequential K halves, +16 for lane>=16
      const _Float16* brow = Bt + (size_t)(n0 + 16 * j + l16) * K + k + 16 * hi;
      const v16h b = *(const v16h*)brow;
      acc[0][j] = wmma_f16(a0.v, b, acc[0][j]);
      acc[1][j] = wmma_f16(a1.v, b, acc[1][j]);
    }
  }

#pragma unroll
  for (int mi = 0; mi < 2; ++mi)
#pragma unroll
    for (int j = 0; j < 4; ++j) {
      const int n = n0 + 16 * j + l16;
      const float bval = bias ? bias[n] : 0.0f;
#pragma unroll
      for (int r = 0; r < 8; ++r) {
        const int m = m0 + 16 * mi + r + 8 * hi;  // C/D layout: lane>=16 -> M=r+8
        C[(size_t)m * N + n] = acc[mi][j][r] + bval;
      }
    }
}

// ---------------------------------------------------------------------------
// Fused RMSNorm (over D=64) + RoPE, f32 in -> f16 out [b][nh][s][d].
// One wave per (b,h,s) row; lane holds d=lane and d=lane+32 so the RoPE
// rotate pair (d, d+32) is lane-local. Block 256 = 8 rows.
// ---------------------------------------------------------------------------
__global__ __launch_bounds__(256)
void rmsnorm_rope_kernel(const float* __restrict__ lin,   // [B*S][NH*64]
                         const float* __restrict__ scale, // [64]
                         const float* __restrict__ cosb,  // [S][64]
                         const float* __restrict__ sinb,  // [S][64]
                         _Float16* __restrict__ outp,     // [B][NH][S][64]
                         int NH) {
  const int lane = threadIdx.x & 31;
  const int wave = threadIdx.x >> 5;
  long row = (long)blockIdx.x * 8 + wave;      // over B*NH*S
  const int s = (int)(row % Sc);
  long t = row / Sc;
  const int h = (int)(t % NH);
  const int b = (int)(t / NH);

  const float* src = lin + (size_t)(b * Sc + s) * (NH * Dc) + h * Dc;
  float x1 = src[lane];
  float x2 = src[lane + 32];

  float ss = x1 * x1 + x2 * x2;
#pragma unroll
  for (int m = 1; m < 32; m <<= 1) ss += __shfl_xor(ss, m, 32);
  const float rs = rsqrtf(ss * (1.0f / 64.0f) + 1e-6f);

  const float n1 = x1 * rs * scale[lane];
  const float n2 = x2 * rs * scale[lane + 32];
  const float* cr = cosb + (size_t)s * Dc;
  const float* sr = sinb + (size_t)s * Dc;
  const float o1 = n1 * cr[lane]      - n2 * sr[lane];        // d < 32
  const float o2 = n2 * cr[lane + 32] + n1 * sr[lane + 32];   // d >= 32

  _Float16* dst = outp + ((size_t)(b * NH + h) * Sc + s) * Dc;
  dst[lane]      = (_Float16)o1;
  dst[lane + 32] = (_Float16)o2;
}

// ---------------------------------------------------------------------------
// Pack V transposed: vT[b][kv][d][s] (f16) = v_lin[(b,s)][kv*64+d] (f32)
// ---------------------------------------------------------------------------
__global__ void pack_vT_kernel(const float* __restrict__ vlin,
                               _Float16* __restrict__ vT) {
  long i = (long)blockIdx.x * blockDim.x + threadIdx.x;  // over B*KV*D*S
  const long total = (long)Bc * KVc * Dc * Sc;
  if (i >= total) return;
  const int s = (int)(i % Sc);
  long t = i / Sc;
  const int d  = (int)(t % Dc);  t /= Dc;
  const int kv = (int)(t % KVc);
  const int b  = (int)(t / KVc);
  vT[i] = (_Float16)vlin[(size_t)(b * Sc + s) * (KVc * Dc) + kv * Dc + d];
}

// ---------------------------------------------------------------------------
// Causal flash attention, WMMA f16, GQA G=4.
// Block = 128 threads = 4 waves; wave owns 16 query rows; 32-key chunks.
// Scores computed TRANSPOSED (S^T = K * Q^T) so the f32 D layout of the
// score tiles is exactly the f16 A-fragment layout required for P*V
// (lane<16: K{0..7,16..23}; lane>=16: K{8..15,24..31}) -> no transpose.
// Online softmax is lane-local + one shfl_xor(16) half-merge per chunk.
// ---------------------------------------------------------------------------
__global__ __launch_bounds__(128)
void flash_attn_wmma_kernel(const _Float16* __restrict__ qh,  // [B][H][S][64]
                            const _Float16* __restrict__ kh,  // [B][KV][S][64]
                            const _Float16* __restrict__ vT,  // [B][KV][64][S]
                            _Float16* __restrict__ ctx) {     // [B*S][H*64]
  const int lane = threadIdx.x & 31;
  const int wave = threadIdx.x >> 5;
  const int hi = lane >> 4, l16 = lane & 15;
  const int b = blockIdx.z, h = blockIdx.y;
  const int kvh = h >> 2;                       // G = 4
  const int q0 = blockIdx.x * 64 + wave * 16;
  const int qrow = q0 + l16;

  const _Float16* qbase = qh + ((size_t)(b * Hc  + h)   * Sc) * Dc;
  const _Float16* kbase = kh + ((size_t)(b * KVc + kvh) * Sc) * Dc;
  const _Float16* vbase = vT + ((size_t)(b * KVc + kvh) * Dc) * Sc;

  // Q as WMMA B-operand: lane = query row l16, halves d = 16*hi + 0..15
  const v16h qf0 = *(const v16h*)(qbase + (size_t)qrow * Dc + 16 * hi);
  const v16h qf1 = *(const v16h*)(qbase + (size_t)qrow * Dc + 32 + 16 * hi);

  v8f o[4] = {};
  float m_prev = -1e30f, lsum = 0.0f;
  const float sm = 0.125f;  // 1/sqrt(64)
  const int kend = q0 + 16; // need keys <= q0+15

  for (int kb = 0; kb < kend; kb += 32) {
    // Speculative prefetch of next chunk (global_prefetch_b8)
    __builtin_prefetch(kbase + (size_t)(kb + 32 + l16) * Dc, 0, 1);
    __builtin_prefetch(vbase + (size_t)l16 * Sc + kb + 32, 0, 1);

    // S^T tiles: rows = keys, cols = queries; accumulate over d = 64
    float sc[16];
#pragma unroll
    for (int t = 0; t < 2; ++t) {
      const _Float16* krow = kbase + (size_t)(kb + 16 * t + l16) * Dc;
      AFrag a0, a1;
      a0.seg[0] = *(const v8h*)(krow + 16 * hi);
      a0.seg[1] = *(const v8h*)(krow + 16 * hi + 8);
      a1.seg[0] = *(const v8h*)(krow + 32 + 16 * hi);
      a1.seg[1] = *(const v8h*)(krow + 40 + 16 * hi);
      v8f c = {};
      c = wmma_f16(a0.v, qf0, c);
      c = wmma_f16(a1.v, qf1, c);
#pragma unroll
      for (int r = 0; r < 8; ++r) sc[8 * t + r] = c[r] * sm;
    }

    // Causal mask: key index = kb + 16*t + r + 8*hi, query = qrow
    if (kb + 31 > q0) {
#pragma unroll
      for (int r = 0; r < 8; ++r) {
        if (kb + r + 8 * hi > qrow)      sc[r]     = -1e30f;
        if (kb + 16 + r + 8 * hi > qrow) sc[8 + r] = -1e30f;
      }
    }

    // Online softmax (lane-local, merge the two key-halves via shfl_xor 16)
    float mloc = sc[0];
#pragma unroll
    for (int i = 1; i < 16; ++i) mloc = fmaxf(mloc, sc[i]);
    mloc = fmaxf(mloc, __shfl_xor(mloc, 16, 32));
    const float m_new = fmaxf(m_prev, mloc);
    const float alpha = __expf(m_prev - m_new);

    AFrag p;  // exp'd scores ARE the P A-fragment (layout identity)
    float rsum = 0.0f;
#pragma unroll
    for (int i = 0; i < 16; ++i) {
      const float e = __expf(sc[i] - m_new);
      rsum += e;
      p.h[i] = (_Float16)e;
    }
    lsum = lsum * alpha + rsum;
    m_prev = m_new;

    // Rescale O accumulators: row M = r + 8*hi -> broadcast alpha per query
    float ab[8];
#pragma unroll
    for (int r = 0; r < 8; ++r) ab[r] = __shfl(alpha, r + 8 * hi, 32);

#pragma unroll
    for (int j = 0; j < 4; ++j) {
#pragma unroll
      for (int r = 0; r < 8; ++r) o[j][r] *= ab[r];
      // V as B-operand: lane = d = 16*j + l16, keys kb + {8*hi+0..7, 16+8*hi+0..7}
      AFrag bv;
      bv.seg[0] = *(const v8h*)(vbase + (size_t)(16 * j + l16) * Sc + kb + 8 * hi);
      bv.seg[1] = *(const v8h*)(vbase + (size_t)(16 * j + l16) * Sc + kb + 16 + 8 * hi);
      o[j] = wmma_f16(p.v, bv.v, o[j]);
    }
  }

  // Normalize and store context (f16, [b][s][h*64+d])
  float lt = lsum + __shfl_xor(lsum, 16, 32);
  const float rl = 1.0f / lt;
  float rb[8];
#pragma unroll
  for (int r = 0; r < 8; ++r) rb[r] = __shfl(rl, r + 8 * hi, 32);

#pragma unroll
  for (int j = 0; j < 4; ++j)
#pragma unroll
    for (int r = 0; r < 8; ++r) {
      const int qq = q0 + r + 8 * hi;
      const int dd = 16 * j + l16;
      ctx[(size_t)(b * Sc + qq) * (Hc * Dc) + h * Dc + dd] =
          (_Float16)(o[j][r] * rb[r]);
    }
}

// ---------------------------------------------------------------------------
// Host launch
// ---------------------------------------------------------------------------
extern "C" void kernel_launch(void* const* d_in, const int* in_sizes, int n_in,
                              void* d_out, int out_size, void* d_ws,
                              size_t ws_size, hipStream_t stream) {
  (void)in_sizes; (void)n_in; (void)out_size; (void)ws_size;
  const float* x    = (const float*)d_in[0];
  // d_in[1] = mask (bool) — causal mask computed analytically
  const float* cosb = (const float*)d_in[2];
  const float* sinb = (const float*)d_in[3];
  const float* Wq   = (const float*)d_in[4];
  const float* bq   = (const float*)d_in[5];
  const float* Wk   = (const float*)d_in[6];
  const float* bk   = (const float*)d_in[7];
  const float* Wv   = (const float*)d_in[8];
  const float* bv   = (const float*)d_in[9];
  const float* Wo   = (const float*)d_in[10];
  const float* qsc  = (const float*)d_in[11];
  const float* ksc  = (const float*)d_in[12];
  float* out = (float*)d_out;

  const size_t M = (size_t)Bc * Sc;          // 4096
  const size_t Nq = (size_t)Hc * Dc;         // 2048
  const size_t Nkv = (size_t)KVc * Dc;       // 512

  char* p = (char*)d_ws;
  auto alloc = [&](size_t bytes) -> char* {
    char* r = p;
    p += (bytes + 255) & ~(size_t)255;
    return r;
  };

  _Float16* xh   = (_Float16*)alloc(M * Ec * 2);
  _Float16* wqT  = (_Float16*)alloc(Nq * Ec * 2);
  _Float16* wkT  = (_Float16*)alloc(Nkv * Ec * 2);
  _Float16* wvT  = (_Float16*)alloc(Nkv * Ec * 2);
  _Float16* woT  = (_Float16*)alloc((size_t)Ec * Nq * 2);
  float*    qlin = (float*)alloc(M * Nq * 4);      // reused as ctx_h later
  float*    klin = (float*)alloc(M * Nkv * 4);
  float*    vlin = (float*)alloc(M * Nkv * 4);
  _Float16* qh   = (_Float16*)alloc(M * Nq * 2);
  _Float16* kh   = (_Float16*)alloc(M * Nkv * 2);
  _Float16* vT   = (_Float16*)alloc(M * Nkv * 2);
  _Float16* ctxh = (_Float16*)qlin;                // alias: qlin dead by then

  // 1) x -> f16
  cvt_f16_kernel<<<4096, 256, 0, stream>>>(x, xh, (long)(M * Ec));

  // 2) weight transposes to [N][K] f16
  transpose_cvt_kernel<<<dim3(Nq / 32, Ec / 32), dim3(32, 8), 0, stream>>>(
      Wq, wqT, Ec, Nq);
  transpose_cvt_kernel<<<dim3(Nkv / 32, Ec / 32), dim3(32, 8), 0, stream>>>(
      Wk, wkT, Ec, Nkv);
  transpose_cvt_kernel<<<dim3(Nkv / 32, Ec / 32), dim3(32, 8), 0, stream>>>(
      Wv, wvT, Ec, Nkv);
  transpose_cvt_kernel<<<dim3(Ec / 32, Nq / 32), dim3(32, 8), 0, stream>>>(
      Wo, woT, Nq, Ec);

  // 3) QKV projections
  gemm_f16_wmma_kernel<<<dim3(Nq / 256, M / 32), 128, 0, stream>>>(
      xh, wqT, bq, qlin, (int)M, (int)Nq, Ec);
  gemm_f16_wmma_kernel<<<dim3(Nkv / 256, M / 32), 128, 0, stream>>>(
      xh, wkT, bk, klin, (int)M, (int)Nkv, Ec);
  gemm_f16_wmma_kernel<<<dim3(Nkv / 256, M / 32), 128, 0, stream>>>(
      xh, wvT, bv, vlin, (int)M, (int)Nkv, Ec);

  // 4) RMSNorm + RoPE (q, k) and V pack
  rmsnorm_rope_kernel<<<(Bc * Hc * Sc) / 8, 256, 0, stream>>>(
      qlin, qsc, cosb, sinb, qh, Hc);
  rmsnorm_rope_kernel<<<(Bc * KVc * Sc) / 8, 256, 0, stream>>>(
      klin, ksc, cosb, sinb, kh, KVc);
  pack_vT_kernel<<<(Bc * KVc * Dc * Sc) / 256, 256, 0, stream>>>(vlin, vT);

  // 5) Flash attention -> ctx (f16), overwrites qlin region
  flash_attn_wmma_kernel<<<dim3(Sc / 64, Hc, Bc), 128, 0, stream>>>(
      qh, kh, vT, ctxh);

  // 6) Output projection: out = ctx @ Wo
  gemm_f16_wmma_kernel<<<dim3(Ec / 256, M / 32), 128, 0, stream>>>(
      ctxh, woT, nullptr, out, (int)M, Ec, (int)Nq);
}
